// ALSDBeamRNNT_46059229282678
// MI455X (gfx1250) — compile-verified, matrix operand order checked by reading, not run on hardware
//
#include <hip/hip_runtime.h>
#include <hip/hip_bf16.h>
#include <math.h>

// ---------------------------------------------------------------------------
// Problem constants (from the reference)
// ---------------------------------------------------------------------------
#define BB     8
#define BEAM   16
#define NN     (BB * BEAM)      // 128
#define TT     128
#define D_ENC  512
#define EE     256
#define HH     512
#define JJ     512
#define VV     4096
#define KCAT   (EE + HH)        // 768
#define FOURH  (4 * HH)         // 2048
#define STEPS  211              // T + int(T*0.65)
#define NEGF   (-1.0e30f)
#define SENT   (-3.0e38f)

typedef __bf16 bf16;
typedef __attribute__((ext_vector_type(16))) __bf16 v16bf;
typedef __attribute__((ext_vector_type(8)))  __bf16 v8bf;
typedef __attribute__((ext_vector_type(8)))  float  v8f;

static __device__ __forceinline__ bf16 f2bf(float f) {
  unsigned u = __builtin_bit_cast(unsigned, f);
  unsigned r = (u + 0x7FFFu + ((u >> 16) & 1u)) >> 16;
  unsigned short s = (unsigned short)r;
  return __builtin_bit_cast(bf16, s);
}
static __device__ __forceinline__ float sigf(float x) { return 1.0f / (1.0f + expf(-x)); }

// Fragment load: 16 bf16 for one wave-striped 16x32 operand tile.
// Lane layout (16-bit A/B 16x32): row/col = lane&15, g = lane>>4;
//   elems 0..7  <- K = k + g*8 .. +7      (one contiguous 16B chunk)
//   elems 8..15 <- K = k + 16 + g*8 .. +7 (one contiguous 16B chunk)
static __device__ __forceinline__ v16bf load_frag(const bf16* __restrict__ base,
                                                  int k, int g8) {
  union { v16bf v; v8bf h[2]; } u;
  u.h[0] = *(const v8bf*)(base + k + g8);
  u.h[1] = *(const v8bf*)(base + k + 16 + g8);
  return u.v;
}

// ---------------------------------------------------------------------------
// One-time prep kernels: fp32 -> bf16 conversions / transposes
// ---------------------------------------------------------------------------
__global__ void k_transpose_bf(const float* __restrict__ in, bf16* __restrict__ out,
                               int K, int NO) {
  int idx = blockIdx.x * blockDim.x + threadIdx.x;
  if (idx >= K * NO) return;
  int o = idx / K, k = idx - o * K;
  out[idx] = f2bf(in[k * NO + o]);
}

// WT_cat[o][k] : o<2048, k<768 ; rows 0..255 from W_ih, 256..767 from W_hh
__global__ void k_cat_transpose_bf(const float* __restrict__ Wih,
                                   const float* __restrict__ Whh,
                                   bf16* __restrict__ out) {
  int idx = blockIdx.x * blockDim.x + threadIdx.x;
  if (idx >= FOURH * KCAT) return;
  int o = idx / KCAT, k = idx - o * KCAT;
  float v = (k < EE) ? Wih[k * FOURH + o] : Whh[(k - EE) * FOURH + o];
  out[idx] = f2bf(v);
}

__global__ void k_convert_bf(const float* __restrict__ in, bf16* __restrict__ out, int n) {
  int idx = blockIdx.x * blockDim.x + threadIdx.x;
  if (idx < n) out[idx] = f2bf(in[idx]);
}

// ---------------------------------------------------------------------------
// WMMA bf16 GEMM:  C[M,Nc] = A[M,K](bf16,row) @ BT[Nc,K](bf16,row)^T + bias
// grid = (Nc/16, M/64), block = 32.
// One wave owns a 64x16 C strip: 4 accumulator tiles share one B (weight)
// fragment per K-chunk (4 WMMAs per B load). K-loop manually unrolled x2 with
// explicit double buffering so next chunk's loads overlap current WMMAs.
// Requires K % 64 == 0 (holds: 768, 512).
// C layout: vgpr rv, lane l -> M = rv + 8*(l>>4), N = l&15.
// ---------------------------------------------------------------------------
__global__ void __launch_bounds__(32) k_gemm_bf16(const bf16* __restrict__ A,
                                                  const bf16* __restrict__ BT,
                                                  const float* __restrict__ bias,
                                                  float* __restrict__ C,
                                                  int Nc, int K) {
  const int lane = threadIdx.x;
  const int g8 = (lane >> 4) << 3;
  const int r = lane & 15;
  const int n0 = blockIdx.x << 4;
  const int m_base = blockIdx.y << 6;

  const bf16* __restrict__ brow = BT + (size_t)(n0 + r) * K;
  const bf16* __restrict__ arow0 = A + (size_t)(m_base + 0 * 16 + r) * K;
  const bf16* __restrict__ arow1 = A + (size_t)(m_base + 1 * 16 + r) * K;
  const bf16* __restrict__ arow2 = A + (size_t)(m_base + 2 * 16 + r) * K;
  const bf16* __restrict__ arow3 = A + (size_t)(m_base + 3 * 16 + r) * K;

  v8f acc0 = {}, acc1 = {}, acc2 = {}, acc3 = {};

  // buffer 0 <- chunk k, buffer 1 <- chunk k+32
  v16bf b0 = load_frag(brow, 0, g8);
  v16bf a00 = load_frag(arow0, 0, g8);
  v16bf a01 = load_frag(arow1, 0, g8);
  v16bf a02 = load_frag(arow2, 0, g8);
  v16bf a03 = load_frag(arow3, 0, g8);
  v16bf b1, a10, a11, a12, a13;

  for (int k = 0; k < K; k += 64) {
    if (k + 256 < K) {
      __builtin_prefetch(brow + k + 256, 0, 1);
      __builtin_prefetch(arow0 + k + 256, 0, 1);
      __builtin_prefetch(arow1 + k + 256, 0, 1);
      __builtin_prefetch(arow2 + k + 256, 0, 1);
      __builtin_prefetch(arow3 + k + 256, 0, 1);
    }
    // preload chunk k+32 into buffer 1, then WMMA on buffer 0
    b1  = load_frag(brow, k + 32, g8);
    a10 = load_frag(arow0, k + 32, g8);
    a11 = load_frag(arow1, k + 32, g8);
    a12 = load_frag(arow2, k + 32, g8);
    a13 = load_frag(arow3, k + 32, g8);
    acc0 = __builtin_amdgcn_wmma_f32_16x16x32_bf16(false, a00, false, b0, (short)0, acc0, false, false);
    acc1 = __builtin_amdgcn_wmma_f32_16x16x32_bf16(false, a01, false, b0, (short)0, acc1, false, false);
    acc2 = __builtin_amdgcn_wmma_f32_16x16x32_bf16(false, a02, false, b0, (short)0, acc2, false, false);
    acc3 = __builtin_amdgcn_wmma_f32_16x16x32_bf16(false, a03, false, b0, (short)0, acc3, false, false);
    // preload chunk k+64 into buffer 0 (if any), then WMMA on buffer 1
    if (k + 64 < K) {
      b0  = load_frag(brow, k + 64, g8);
      a00 = load_frag(arow0, k + 64, g8);
      a01 = load_frag(arow1, k + 64, g8);
      a02 = load_frag(arow2, k + 64, g8);
      a03 = load_frag(arow3, k + 64, g8);
    }
    acc0 = __builtin_amdgcn_wmma_f32_16x16x32_bf16(false, a10, false, b1, (short)0, acc0, false, false);
    acc1 = __builtin_amdgcn_wmma_f32_16x16x32_bf16(false, a11, false, b1, (short)0, acc1, false, false);
    acc2 = __builtin_amdgcn_wmma_f32_16x16x32_bf16(false, a12, false, b1, (short)0, acc2, false, false);
    acc3 = __builtin_amdgcn_wmma_f32_16x16x32_bf16(false, a13, false, b1, (short)0, acc3, false, false);
  }

  const float bv = bias ? bias[n0 + r] : 0.0f;
  const int gq = g8;  // row offset within tile for this half-wave
#pragma unroll
  for (int rv = 0; rv < 8; ++rv) {
    C[(size_t)(m_base +  0 + rv + gq) * Nc + (n0 + r)] = acc0[rv] + bv;
    C[(size_t)(m_base + 16 + rv + gq) * Nc + (n0 + r)] = acc1[rv] + bv;
    C[(size_t)(m_base + 32 + rv + gq) * Nc + (n0 + r)] = acc2[rv] + bv;
    C[(size_t)(m_base + 48 + rv + gq) * Nc + (n0 + r)] = acc3[rv] + bv;
  }
}

// ---------------------------------------------------------------------------
// Per-step kernels
// ---------------------------------------------------------------------------
__global__ void k_pack_xh(const float* __restrict__ embed, const int* __restrict__ cur_tok,
                          const float* __restrict__ h, bf16* __restrict__ xh) {
  int idx = blockIdx.x * blockDim.x + threadIdx.x;
  if (idx >= NN * KCAT) return;
  int n = idx / KCAT, k = idx - n * KCAT;
  float v = (k < EE) ? embed[(size_t)cur_tok[n] * EE + k] : h[n * HH + (k - EE)];
  xh[idx] = f2bf(v);
}

// LSTM elementwise on z[N,4H] (bias already added in GEMM)
__global__ void k_lstm(const float* __restrict__ z, const float* __restrict__ c_old,
                       float* __restrict__ h2, float* __restrict__ c2,
                       bf16* __restrict__ h2bf) {
  int idx = blockIdx.x * blockDim.x + threadIdx.x;
  if (idx >= NN * HH) return;
  int n = idx / HH, j = idx - n * HH;
  const float* zr = z + (size_t)n * FOURH;
  float i_ = sigf(zr[j]);
  float f_ = sigf(zr[HH + j]);
  float g_ = tanhf(zr[2 * HH + j]);
  float o_ = sigf(zr[3 * HH + j]);
  float c2v = f_ * c_old[idx] + i_ * g_;
  float h2v = o_ * tanhf(c2v);
  c2[idx] = c2v;
  h2[idx] = h2v;
  h2bf[idx] = f2bf(h2v);
}

// Per-row step metadata: t index into encoder, skip/end flag
__global__ void k_meta(const int* __restrict__ enc_len, const int* __restrict__ seq_len,
                       int step, int* __restrict__ t_idx, int* __restrict__ flag) {
  int n = threadIdx.x;
  if (n >= NN) return;
  int b = n / BEAM;
  int L = enc_len[b];
  int umax = (int)((float)L * 0.65f);
  int total = umax + L;
  int i = (step < total - 1) ? step : (total - 1);
  int ends = (step >= total - 1) ? 1 : 0;
  int t = i - seq_len[n];
  int skips = (t > L - 1) ? 1 : 0;
  if (skips) t = 0;
  if (t < 0) t = 0;
  if (t > TT - 1) t = TT - 1;
  t_idx[n] = t;
  flag[n] = skips | ends;
}

// joint_bf[n][j] = bf16(tanh(enc_proj[b*T+t][j] + dec_proj[n][j]))
__global__ void k_joint(const float* __restrict__ enc_proj, const float* __restrict__ dec_proj,
                        const int* __restrict__ t_idx, bf16* __restrict__ joint_bf) {
  int idx = blockIdx.x * blockDim.x + threadIdx.x;
  if (idx >= NN * JJ) return;
  int n = idx / JJ, j = idx - n * JJ;
  int b = n / BEAM;
  size_t row = (size_t)(b * TT + t_idx[n]) * JJ;
  joint_bf[idx] = f2bf(tanhf(enc_proj[row + j] + dec_proj[idx]));
}

// In-place log-softmax over V per row; one block per row
__global__ void __launch_bounds__(256) k_logsoftmax(float* __restrict__ logits) {
  __shared__ float red[256];
  int n = blockIdx.x, tid = threadIdx.x;
  float* row = logits + (size_t)n * VV;
  float m = SENT;
  for (int i = tid; i < VV; i += 256) m = fmaxf(m, row[i]);
  red[tid] = m; __syncthreads();
  for (int off = 128; off > 0; off >>= 1) {
    if (tid < off) red[tid] = fmaxf(red[tid], red[tid + off]);
    __syncthreads();
  }
  m = red[0]; __syncthreads();
  float s = 0.0f;
  for (int i = tid; i < VV; i += 256) s += expf(row[i] - m);
  red[tid] = s; __syncthreads();
  for (int off = 128; off > 0; off >>= 1) {
    if (tid < off) red[tid] += red[tid + off];
    __syncthreads();
  }
  float lse = m + logf(red[0]);
  for (int i = tid; i < VV; i += 256) row[i] -= lse;
}

// Per-row top-(BEAM-1) over cols 1..V-1 plus blank col handling.
// Iterative argmax, lower-index tie-break (matches jax.lax.top_k ordering).
__global__ void __launch_bounds__(256) k_topk(const float* __restrict__ logp,
                                              const int* __restrict__ flag,
                                              float* __restrict__ vals, int* __restrict__ ids) {
  __shared__ float sv[256];
  __shared__ int   si[256];
  int n = blockIdx.x, tid = threadIdx.x;
  const float* row = logp + (size_t)n * VV;
  float lv[16];
#pragma unroll
  for (int e = 0; e < 16; ++e) lv[e] = row[tid * 16 + e];
  int fl = flag[n];
  if (tid == 0) {
    vals[n * BEAM] = fl ? 0.0f : lv[0];  // blank column (frozen -> 0)
    ids[n * BEAM] = 0;
    lv[0] = SENT;                         // exclude blank from top-k over [1:]
  }
  for (int s = 1; s < BEAM; ++s) {
    float bestv = SENT; int besti = VV;
#pragma unroll
    for (int e = 0; e < 16; ++e)
      if (lv[e] > bestv) { bestv = lv[e]; besti = tid * 16 + e; }
    sv[tid] = bestv; si[tid] = besti; __syncthreads();
    for (int off = 128; off > 0; off >>= 1) {
      if (tid < off) {
        float v2 = sv[tid + off]; int i2 = si[tid + off];
        if (v2 > sv[tid] || (v2 == sv[tid] && i2 < si[tid])) { sv[tid] = v2; si[tid] = i2; }
      }
      __syncthreads();
    }
    int bi = si[0]; float bv = sv[0];
    if (tid == 0) {
      vals[n * BEAM + s] = fl ? NEGF : bv;
      ids[n * BEAM + s] = bi;              // absolute column == idx+1 in reference
    }
#pragma unroll
    for (int e = 0; e < 16; ++e)
      if (tid * 16 + e == bi) lv[e] = SENT;
    __syncthreads();
  }
}

// Per-batch beam merge over BEAM*BEAM candidates; top-BEAM with tie-break.
__global__ void __launch_bounds__(256) k_beam(const float* __restrict__ scores_old,
                                              const float* __restrict__ vals,
                                              const int* __restrict__ ids,
                                              float* __restrict__ scores_new,
                                              int* __restrict__ parent,
                                              int* __restrict__ next_tok) {
  __shared__ float sv[256];
  __shared__ int   si[256];
  int b = blockIdx.x, tid = threadIdx.x;
  int j = tid >> 4, k = tid & 15;
  int row = b * BEAM + j;
  float cand = scores_old[row] + vals[row * BEAM + k];
  for (int s = 0; s < BEAM; ++s) {
    sv[tid] = cand; si[tid] = tid; __syncthreads();
    for (int off = 128; off > 0; off >>= 1) {
      if (tid < off) {
        float v2 = sv[tid + off]; int i2 = si[tid + off];
        if (v2 > sv[tid] || (v2 == sv[tid] && i2 < si[tid])) { sv[tid] = v2; si[tid] = i2; }
      }
      __syncthreads();
    }
    int bi = si[0]; float bv = sv[0];
    if (tid == 0) {
      int p = b * BEAM + (bi >> 4);
      scores_new[b * BEAM + s] = bv;
      parent[b * BEAM + s] = p;
      next_tok[b * BEAM + s] = ids[p * BEAM + (bi & 15)];
    }
    if (tid == bi) cand = SENT;
    __syncthreads();
  }
}

// Gather parent state, write hypotheses, maintain last_tok/seq_len incrementally.
__global__ void __launch_bounds__(256) k_update(const int* __restrict__ parent,
                                                const int* __restrict__ next_tok,
                                                const int* __restrict__ flag,
                                                const int* __restrict__ hyps_old,
                                                int* __restrict__ hyps_new,
                                                const float* __restrict__ h_old,
                                                const float* __restrict__ c_old,
                                                const float* __restrict__ h2,
                                                const float* __restrict__ c2,
                                                float* __restrict__ h_new,
                                                float* __restrict__ c_new,
                                                const int* __restrict__ seq_old,
                                                int* __restrict__ seq_new,
                                                const int* __restrict__ last_old,
                                                int* __restrict__ last_new,
                                                int* __restrict__ cur_tok, int step) {
  int n = blockIdx.x, tid = threadIdx.x;
  int p = parent[n];
  int tok = next_tok[n];
  int fl = flag[p];
  int hyp_tok = fl ? 0 : tok;
  int is_blank = (tok == 0);
  for (int i = tid; i < STEPS; i += 256) {
    int v = hyps_old[p * STEPS + i];
    if (i == step) v = hyp_tok;
    hyps_new[n * STEPS + i] = v;
  }
  const float* hs = is_blank ? h_old : h2;
  const float* cs = is_blank ? c_old : c2;
  for (int j = tid; j < HH; j += 256) {
    h_new[n * HH + j] = hs[p * HH + j];
    c_new[n * HH + j] = cs[p * HH + j];
  }
  if (tid == 0) {
    int lt = last_old[p];
    cur_tok[n] = is_blank ? lt : tok;
    last_new[n] = hyp_tok ? hyp_tok : lt;
    seq_new[n] = seq_old[p] + (hyp_tok ? 1 : 0);
  }
}

__global__ void k_init(float* __restrict__ scores, int* __restrict__ hyps,
                       int* __restrict__ cur_tok, float* __restrict__ h,
                       float* __restrict__ c, int* __restrict__ seq,
                       int* __restrict__ last) {
  int idx = blockIdx.x * blockDim.x + threadIdx.x;
  if (idx < NN * HH) { h[idx] = 0.0f; c[idx] = 0.0f; }
  if (idx < NN * STEPS) hyps[idx] = 0;
  if (idx < NN) {
    scores[idx] = (idx % BEAM == 0) ? 0.0f : NEGF;
    cur_tok[idx] = 0; seq[idx] = 0; last[idx] = 0;
  }
}

// d_out = [hyps int32 (N*STEPS)] ++ [scores f32 (N)]
__global__ void k_out(const int* __restrict__ hyps, const float* __restrict__ scores,
                      void* __restrict__ out) {
  int idx = blockIdx.x * blockDim.x + threadIdx.x;
  if (idx < NN * STEPS) ((int*)out)[idx] = hyps[idx];
  if (idx < NN) ((float*)out)[NN * STEPS + idx] = scores[idx];
}

// ---------------------------------------------------------------------------
// Host orchestration
// ---------------------------------------------------------------------------
extern "C" void kernel_launch(void* const* d_in, const int* in_sizes, int n_in,
                              void* d_out, int out_size, void* d_ws, size_t ws_size,
                              hipStream_t stream) {
  (void)in_sizes; (void)n_in; (void)out_size; (void)ws_size;
  const float* enc_out = (const float*)d_in[0];
  const int*   enc_len = (const int*)d_in[1];
  const float* embed   = (const float*)d_in[2];
  const float* W_ih    = (const float*)d_in[3];
  const float* W_hh    = (const float*)d_in[4];
  const float* b_lstm  = (const float*)d_in[5];
  const float* W_enc   = (const float*)d_in[6];
  const float* W_dec   = (const float*)d_in[7];
  const float* b_joint = (const float*)d_in[8];
  const float* W_out   = (const float*)d_in[9];
  const float* b_out   = (const float*)d_in[10];

  char* ws = (char*)d_ws;
  size_t off = 0;
  auto alloc = [&](size_t bytes) -> char* {
    char* p = ws + off;
    off = (off + bytes + 255) & ~(size_t)255;
    return p;
  };

  bf16*  wt_cat   = (bf16*)alloc((size_t)FOURH * KCAT * 2);
  bf16*  wt_dec   = (bf16*)alloc((size_t)JJ * HH * 2);
  bf16*  wt_out   = (bf16*)alloc((size_t)VV * JJ * 2);
  bf16*  wt_enc   = (bf16*)alloc((size_t)JJ * D_ENC * 2);
  bf16*  enc_bf   = (bf16*)alloc((size_t)BB * TT * D_ENC * 2);
  float* enc_proj = (float*)alloc((size_t)BB * TT * JJ * 4);
  bf16*  xh       = (bf16*)alloc((size_t)NN * KCAT * 2);
  float* z        = (float*)alloc((size_t)NN * FOURH * 4);
  float* h2       = (float*)alloc((size_t)NN * HH * 4);
  float* c2       = (float*)alloc((size_t)NN * HH * 4);
  bf16*  h2bf     = (bf16*)alloc((size_t)NN * HH * 2);
  float* dec_proj = (float*)alloc((size_t)NN * JJ * 4);
  bf16*  joint_bf = (bf16*)alloc((size_t)NN * JJ * 2);
  float* logits   = (float*)alloc((size_t)NN * VV * 4);
  float* vals     = (float*)alloc((size_t)NN * BEAM * 4);
  int*   ids      = (int*)alloc((size_t)NN * BEAM * 4);
  int*   t_idx    = (int*)alloc(NN * 4);
  int*   flag     = (int*)alloc(NN * 4);
  int*   parent   = (int*)alloc(NN * 4);
  int*   next_tok = (int*)alloc(NN * 4);
  int*   cur_tok  = (int*)alloc(NN * 4);
  float* scoresA  = (float*)alloc(NN * 4);
  float* scoresB  = (float*)alloc(NN * 4);
  int*   seqA     = (int*)alloc(NN * 4);
  int*   seqB     = (int*)alloc(NN * 4);
  int*   lastA    = (int*)alloc(NN * 4);
  int*   lastB    = (int*)alloc(NN * 4);
  float* hA       = (float*)alloc((size_t)NN * HH * 4);
  float* hB       = (float*)alloc((size_t)NN * HH * 4);
  float* cA       = (float*)alloc((size_t)NN * HH * 4);
  float* cB       = (float*)alloc((size_t)NN * HH * 4);
  int*   hypsA    = (int*)alloc((size_t)NN * STEPS * 4);
  int*   hypsB    = (int*)alloc((size_t)NN * STEPS * 4);

  const int TB = 256;
  // --- one-time weight prep ---
  k_cat_transpose_bf<<<(FOURH * KCAT + TB - 1) / TB, TB, 0, stream>>>(W_ih, W_hh, wt_cat);
  k_transpose_bf<<<(HH * JJ + TB - 1) / TB, TB, 0, stream>>>(W_dec, wt_dec, HH, JJ);
  k_transpose_bf<<<(JJ * VV + TB - 1) / TB, TB, 0, stream>>>(W_out, wt_out, JJ, VV);
  k_transpose_bf<<<(D_ENC * JJ + TB - 1) / TB, TB, 0, stream>>>(W_enc, wt_enc, D_ENC, JJ);
  k_convert_bf<<<(BB * TT * D_ENC + TB - 1) / TB, TB, 0, stream>>>(enc_out, enc_bf, BB * TT * D_ENC);
  // enc_proj = enc_bf @ W_enc + b_joint   [1024, 512]
  k_gemm_bf16<<<dim3(JJ / 16, (BB * TT) / 64), 32, 0, stream>>>(enc_bf, wt_enc, b_joint, enc_proj, JJ, D_ENC);
  // --- init beam state ---
  k_init<<<(NN * HH + TB - 1) / TB, TB, 0, stream>>>(scoresA, hypsA, cur_tok, hA, cA, seqA, lastA);

  float* h_cur = hA;  float* h_nxt = hB;
  float* c_cur = cA;  float* c_nxt = cB;
  float* sc_cur = scoresA; float* sc_nxt = scoresB;
  int* seq_cur = seqA; int* seq_nxt = seqB;
  int* last_cur = lastA; int* last_nxt = lastB;
  int* hy_cur = hypsA; int* hy_nxt = hypsB;

  for (int step = 0; step < STEPS; ++step) {
    k_meta<<<1, NN, 0, stream>>>(enc_len, seq_cur, step, t_idx, flag);
    k_pack_xh<<<(NN * KCAT + TB - 1) / TB, TB, 0, stream>>>(embed, cur_tok, h_cur, xh);
    // z = [x|h] @ [W_ih;W_hh] + b_lstm   [128, 2048]
    k_gemm_bf16<<<dim3(FOURH / 16, NN / 64), 32, 0, stream>>>(xh, wt_cat, b_lstm, z, FOURH, KCAT);
    k_lstm<<<(NN * HH + TB - 1) / TB, TB, 0, stream>>>(z, c_cur, h2, c2, h2bf);
    // dec_proj = h2 @ W_dec   [128, 512]
    k_gemm_bf16<<<dim3(JJ / 16, NN / 64), 32, 0, stream>>>(h2bf, wt_dec, (const float*)nullptr, dec_proj, JJ, HH);
    k_joint<<<(NN * JJ + TB - 1) / TB, TB, 0, stream>>>(enc_proj, dec_proj, t_idx, joint_bf);
    // logits = joint @ W_out + b_out   [128, 4096]
    k_gemm_bf16<<<dim3(VV / 16, NN / 64), 32, 0, stream>>>(joint_bf, wt_out, b_out, logits, VV, JJ);
    k_logsoftmax<<<NN, 256, 0, stream>>>(logits);
    k_topk<<<NN, 256, 0, stream>>>(logits, flag, vals, ids);
    k_beam<<<BB, 256, 0, stream>>>(sc_cur, vals, ids, sc_nxt, parent, next_tok);
    k_update<<<NN, 256, 0, stream>>>(parent, next_tok, flag, hy_cur, hy_nxt,
                                     h_cur, c_cur, h2, c2, h_nxt, c_nxt,
                                     seq_cur, seq_nxt, last_cur, last_nxt, cur_tok, step);
    // swap ping-pong buffers
    float* tf;
    tf = h_cur;  h_cur = h_nxt;  h_nxt = tf;
    tf = c_cur;  c_cur = c_nxt;  c_nxt = tf;
    tf = sc_cur; sc_cur = sc_nxt; sc_nxt = tf;
    int* ti;
    ti = seq_cur;  seq_cur = seq_nxt;  seq_nxt = ti;
    ti = last_cur; last_cur = last_nxt; last_nxt = ti;
    ti = hy_cur;   hy_cur = hy_nxt;    hy_nxt = ti;
  }

  k_out<<<(NN * STEPS + TB - 1) / TB, TB, 0, stream>>>(hy_cur, sc_cur, d_out);
}
